// CrossAttention_75222057222335
// MI455X (gfx1250) — compile-verified, matrix operand order checked by reading, not run on hardware
//
#include <hip/hip_runtime.h>
#include <hip/hip_bf16.h>

// ---------------------------------------------------------------------------
// CrossAttention for MI455X (gfx1250): LN -> QKV bf16 WMMA GEMMs -> flash
// attention (WMMA for QK^T and PV, online softmax, TDM Q-tile stage) -> LN.
// All matmuls use v_wmma_f32_16x16x32_bf16 (f32 accumulate).
// ---------------------------------------------------------------------------

typedef __attribute__((ext_vector_type(16))) __bf16 v16bf;
typedef __attribute__((ext_vector_type(8)))  float  v8f;

#define D_MODEL 1024
#define B_SZ    32
#define LT      1024
#define LA      2048

// Tensor Data Mover path: builtin exists on both toolchains, arity differs
// (clang-22/ROCm7.2: 5 args, clang-23/amdgpu-toolchain: 6 args).
#if defined(__AMDGCN__) && __has_builtin(__builtin_amdgcn_tensor_load_to_lds) && \
    __has_builtin(__builtin_amdgcn_s_wait_tensorcnt)
#define USE_TDM 1
#endif

typedef unsigned int v4u __attribute__((ext_vector_type(4)));
typedef int          v8i __attribute__((ext_vector_type(8)));
typedef int          v4i __attribute__((ext_vector_type(4)));

// float -> bf16 with round-to-nearest-even
static __device__ __forceinline__ unsigned short f2bf(float f) {
  unsigned u = __float_as_uint(f);
  unsigned r = u + 0x7FFFu + ((u >> 16) & 1u);
  return (unsigned short)(r >> 16);
}

// Load one 16-bit WMMA A/B fragment for this lane.
// Per ISA 16-bit layout: lane holds 8 contiguous K-halves at (k0 + hi8) and
// 8 more at (k0 + hi8 + 16), where hi8 = (lane>=16)*8.  Two b128 loads.
static __device__ __forceinline__ v16bf ldfrag(const unsigned short* p, int hi8) {
  p += hi8;
  uint4 x = *reinterpret_cast<const uint4*>(p);
  uint4 y = *reinterpret_cast<const uint4*>(p + 16);
  v16bf v;
  __builtin_memcpy(&v, &x, 16);
  __builtin_memcpy(reinterpret_cast<char*>(&v) + 16, &y, 16);
  return v;
}

// ---------------------------------------------------------------------------
// LayerNorm (wave per row of 1024) -> bf16 out
// ---------------------------------------------------------------------------
__global__ __launch_bounds__(256) void ln_to_bf16(
    const float* __restrict__ x, const float* __restrict__ g,
    const float* __restrict__ be, unsigned short* __restrict__ y, int rows) {
  int wid = threadIdx.x >> 5, lane = threadIdx.x & 31;
  int row = blockIdx.x * 8 + wid;
  if (row >= rows) return;
  const float4* xr =
      reinterpret_cast<const float4*>(x + (size_t)row * D_MODEL) + lane * 8;
  float4 v[8];
  float s = 0.f, s2 = 0.f;
#pragma unroll
  for (int i = 0; i < 8; ++i) {
    v[i] = xr[i];
    s  += v[i].x + v[i].y + v[i].z + v[i].w;
    s2 += v[i].x * v[i].x + v[i].y * v[i].y + v[i].z * v[i].z + v[i].w * v[i].w;
  }
#pragma unroll
  for (int off = 16; off > 0; off >>= 1) {
    s  += __shfl_xor(s, off, 32);
    s2 += __shfl_xor(s2, off, 32);
  }
  float mean = s * (1.0f / D_MODEL);
  float var  = s2 * (1.0f / D_MODEL) - mean * mean;
  float rstd = rsqrtf(var + 1e-5f);
  int c0 = lane * 32;
  unsigned short* yr = y + (size_t)row * D_MODEL + c0;
#pragma unroll
  for (int i = 0; i < 8; ++i) {
    float vv[4] = {v[i].x, v[i].y, v[i].z, v[i].w};
#pragma unroll
    for (int j = 0; j < 4; ++j) {
      int c = c0 + i * 4 + j;
      yr[i * 4 + j] = f2bf((vv[j] - mean) * rstd * g[c] + be[c]);
    }
  }
}

// LayerNorm -> f32 out (final projection LN)
__global__ __launch_bounds__(256) void ln_to_f32(
    const float* __restrict__ x, const float* __restrict__ g,
    const float* __restrict__ be, float* __restrict__ y, int rows) {
  int wid = threadIdx.x >> 5, lane = threadIdx.x & 31;
  int row = blockIdx.x * 8 + wid;
  if (row >= rows) return;
  const float4* xr =
      reinterpret_cast<const float4*>(x + (size_t)row * D_MODEL) + lane * 8;
  float4 v[8];
  float s = 0.f, s2 = 0.f;
#pragma unroll
  for (int i = 0; i < 8; ++i) {
    v[i] = xr[i];
    s  += v[i].x + v[i].y + v[i].z + v[i].w;
    s2 += v[i].x * v[i].x + v[i].y * v[i].y + v[i].z * v[i].z + v[i].w * v[i].w;
  }
#pragma unroll
  for (int off = 16; off > 0; off >>= 1) {
    s  += __shfl_xor(s, off, 32);
    s2 += __shfl_xor(s2, off, 32);
  }
  float mean = s * (1.0f / D_MODEL);
  float var  = s2 * (1.0f / D_MODEL) - mean * mean;
  float rstd = rsqrtf(var + 1e-5f);
  int c0 = lane * 32;
  float* yr = y + (size_t)row * D_MODEL + c0;
#pragma unroll
  for (int i = 0; i < 8; ++i) {
    float vv[4] = {v[i].x, v[i].y, v[i].z, v[i].w};
#pragma unroll
    for (int j = 0; j < 4; ++j) {
      int c = c0 + i * 4 + j;
      yr[i * 4 + j] = (vv[j] - mean) * rstd * g[c] + be[c];
    }
  }
}

// ---------------------------------------------------------------------------
// Weight cast + transpose: Wt[n][k] = bf16(W[k][n])   (D x D)
// ---------------------------------------------------------------------------
__global__ __launch_bounds__(256) void cast_transpose(
    const float* __restrict__ w, unsigned short* __restrict__ wt) {
  int idx = blockIdx.x * 256 + threadIdx.x;  // idx = n*D + k
  int n = idx >> 10, k = idx & 1023;
  wt[idx] = f2bf(w[k * D_MODEL + n]);
}

// ---------------------------------------------------------------------------
// bf16 GEMM: C[M][N] = A[M][K] @ Bt[N][K]^T + bias, f32 accumulate via WMMA.
// Block: 128x128 tile, 8 waves as 4(M) x 2(N), wave tile 32x64 (2x4 frags).
// transposed==1: write C as [batch][N][Lb] (V stored pre-transposed for PV).
// ---------------------------------------------------------------------------
__global__ __launch_bounds__(256) void gemm_bf16(
    const unsigned short* __restrict__ A, const unsigned short* __restrict__ Bt,
    const float* __restrict__ bias, unsigned short* __restrict__ C,
    int M, int N, int K, int transposed, int Lb) {
  int tid = threadIdx.x, lane = tid & 31, wid = tid >> 5;
  int hi8 = (lane >> 4) << 3;
  int n16 = lane & 15;
  int m0 = blockIdx.y * 128 + (wid & 3) * 32;
  int n0 = blockIdx.x * 128 + (wid >> 2) * 64;

  const unsigned short* arow[2];
  const unsigned short* brow[4];
#pragma unroll
  for (int i = 0; i < 2; ++i) arow[i] = A + (size_t)(m0 + i * 16 + n16) * K;
#pragma unroll
  for (int j = 0; j < 4; ++j) brow[j] = Bt + (size_t)(n0 + j * 16 + n16) * K;

  v8f zero = {};
  v8f acc[2][4];
#pragma unroll
  for (int i = 0; i < 2; ++i)
#pragma unroll
    for (int j = 0; j < 4; ++j) acc[i][j] = zero;

  for (int kk = 0; kk < K; kk += 32) {
    // prefetch one K-step ahead (lowers to global_prefetch_b8)
    if (kk + 32 < K) {
      __builtin_prefetch(arow[0] + kk + 32, 0, 1);
      __builtin_prefetch(brow[0] + kk + 32, 0, 1);
    }
    v16bf af[2], bf[4];
#pragma unroll
    for (int i = 0; i < 2; ++i) af[i] = ldfrag(arow[i] + kk, hi8);
#pragma unroll
    for (int j = 0; j < 4; ++j) bf[j] = ldfrag(brow[j] + kk, hi8);
#pragma unroll
    for (int i = 0; i < 2; ++i)
#pragma unroll
      for (int j = 0; j < 4; ++j)
        acc[i][j] = __builtin_amdgcn_wmma_f32_16x16x32_bf16(
            false, af[i], false, bf[j], (short)0, acc[i][j], false, false);
  }

#pragma unroll
  for (int i = 0; i < 2; ++i)
#pragma unroll
    for (int j = 0; j < 4; ++j) {
      int n = n0 + j * 16 + n16;
      float bs = bias[n];
#pragma unroll
      for (int r = 0; r < 8; ++r) {
        int m = m0 + i * 16 + r + hi8;  // C/D layout: M = r + (lane>=16)*8
        float val = acc[i][j][r] + bs;
        if (!transposed) {
          C[(size_t)m * N + n] = f2bf(val);
        } else {
          int bt = m / Lb, sidx = m % Lb;
          C[((size_t)bt * N + n) * Lb + sidx] = f2bf(val);
        }
      }
    }
}

// ---------------------------------------------------------------------------
// Flash attention: block = (batch, 16-query tile), 8 waves.
// Q tile staged to LDS via the Tensor Data Mover (TDM); 8 chunks of 256 keys;
// online softmax; per-wave 128 D-cols.
// ---------------------------------------------------------------------------
__global__ __launch_bounds__(256) void flash_attn(
    const unsigned short* __restrict__ Q,    // [B*Lt][D]
    const unsigned short* __restrict__ Kb,   // [B*La][D]
    const unsigned short* __restrict__ Vt,   // [B][D][La]
    const int* __restrict__ mask,            // [B][La]
    float* __restrict__ Oo) {                // [B*Lt][D]
  __shared__ unsigned short qs[16 * D_MODEL];  // 32 KB
  __shared__ float ss[16 * 256];               // 16 KB scores chunk
  __shared__ unsigned short ps[16 * 256];      // 8 KB  P (bf16) chunk
  __shared__ float m_run[16], l_run[16], sc_s[16];

  int tid = threadIdx.x, lane = tid & 31, wid = tid >> 5;
  int n16 = lane & 15, hi8 = (lane >> 4) << 3;
  int b  = blockIdx.x >> 6;  // Lt/16 == 64 tiles per batch
  int qt = blockIdx.x & 63;

  // ---- Stage Q tile: one contiguous 32 KB block = trivial 1-D TDM tile ----
#ifdef USE_TDM
  if (wid == 0) {
    unsigned long long ga =
        (unsigned long long)(size_t)(Q + (size_t)(b * LT + qt * 16) * D_MODEL);
    unsigned lds_addr = (unsigned)(size_t)qs;
    // D# group0: count=1 | lds_addr | global_addr[56:0] | type=2
    v4u g0 = {1u, lds_addr, (unsigned)(ga & 0xFFFFFFFFu),
              (unsigned)(((ga >> 32) & 0x01FFFFFFu) | (2u << 30))};
    // D# group1: data_size=3 (8B); tensor_dim0=4096; tensor_dim1=1;
    //            tile_dim0=4096 (1-D tile); tensor_dim0_stride=4096
    v8i g1 = {(int)(3u << 16),
              (int)(4096u << 16),  // tensor_dim0[15:0] @ bits 63:48
              (int)(1u << 16),     // tensor_dim0[31:16]=0 | tensor_dim1[15:0]=1
              (int)(4096u << 16),  // tensor_dim1[31:16]=0 | tile_dim0=4096
              0,                   // tile_dim1=0, tile_dim2=0 (unused)
              4096,                // tensor_dim0_stride[31:0]
              0, 0};               // stride hi bits, dim1_stride = 0
    v4i g2 = {0, 0, 0, 0};
    v4i g3 = {0, 0, 0, 0};
#if defined(__clang_major__) && (__clang_major__ >= 23)
    v8i gx = {0, 0, 0, 0, 0, 0, 0, 0};
    __builtin_amdgcn_tensor_load_to_lds(g0, g1, g2, g3, gx, 0);
#else
    __builtin_amdgcn_tensor_load_to_lds(g0, g1, g2, g3, 0);
#endif
    __builtin_amdgcn_s_wait_tensorcnt(0);
  }
#else
  {
    const uint4* qsrc = reinterpret_cast<const uint4*>(
        Q + (size_t)(b * LT + qt * 16) * D_MODEL);
    for (int i = tid; i < 16 * D_MODEL / 8; i += 256)
      reinterpret_cast<uint4*>(qs)[i] = qsrc[i];
  }
#endif

  if (tid < 16) { m_run[tid] = -1e30f; l_run[tid] = 0.f; }
  v8f zero = {};
  v8f O[8];
#pragma unroll
  for (int i = 0; i < 8; ++i) O[i] = zero;
  __syncthreads();

  int dbase = wid * 128;

  for (int c = 0; c < 8; ++c) {
    int key0 = c * 256;

    // ---- scores S = (Q K^T) * scale, masked; 2 key tiles per wave ----
    for (int t = wid; t < 16; t += 8) {
      int key = key0 + t * 16 + n16;
      const unsigned short* krow = Kb + (size_t)(b * LA + key) * D_MODEL;
      int mk = mask[b * LA + key];
      v8f acc = zero;
      for (int kk = 0; kk < D_MODEL; kk += 32) {
        v16bf aq = ldfrag(qs + n16 * D_MODEL + kk, hi8);
        v16bf bk = ldfrag(krow + kk, hi8);
        acc = __builtin_amdgcn_wmma_f32_16x16x32_bf16(
            false, aq, false, bk, (short)0, acc, false, false);
      }
#pragma unroll
      for (int r = 0; r < 8; ++r) {
        float sv = acc[r] * 0.03125f;  // 1/sqrt(1024)
        if (mk == 0) sv = -1e9f;
        ss[(r + hi8) * 256 + t * 16 + n16] = sv;
      }
    }
    __syncthreads();

    // ---- online softmax row update ----
    if (tid < 16) {
      int r = tid;
      float mc = -1e30f;
      for (int j = 0; j < 256; ++j) mc = fmaxf(mc, ss[r * 256 + j]);
      float mo = m_run[r];
      float mn = fmaxf(mo, mc);
      float sc = __expf(mo - mn);
      float sum = 0.f;
      for (int j = 0; j < 256; ++j) {
        float p = __expf(ss[r * 256 + j] - mn);
        sum += p;
        ps[r * 256 + j] = f2bf(p);
      }
      m_run[r] = mn;
      l_run[r] = l_run[r] * sc + sum;
      sc_s[r]  = sc;
    }
    __syncthreads();

    // ---- rescale running O, then O += P @ Vchunk ----
    float scl[8];
#pragma unroll
    for (int r = 0; r < 8; ++r) scl[r] = sc_s[r + hi8];
#pragma unroll
    for (int nf = 0; nf < 8; ++nf)
#pragma unroll
      for (int r = 0; r < 8; ++r) O[nf][r] *= scl[r];

    for (int kc = 0; kc < 8; ++kc) {
      v16bf ap = ldfrag(ps + n16 * 256 + kc * 32, hi8);
#pragma unroll
      for (int nf = 0; nf < 8; ++nf) {
        int dcol = dbase + nf * 16 + n16;
        const unsigned short* vrow =
            Vt + ((size_t)b * D_MODEL + dcol) * LA + key0 + kc * 32;
        v16bf bv = ldfrag(vrow, hi8);
        O[nf] = __builtin_amdgcn_wmma_f32_16x16x32_bf16(
            false, ap, false, bv, (short)0, O[nf], false, false);
      }
    }
  }

  float li[8];
#pragma unroll
  for (int r = 0; r < 8; ++r) li[r] = 1.0f / l_run[r + hi8];
#pragma unroll
  for (int nf = 0; nf < 8; ++nf) {
    int col = dbase + nf * 16 + n16;
#pragma unroll
    for (int r = 0; r < 8; ++r) {
      int m = qt * 16 + r + hi8;
      Oo[(size_t)(b * LT + m) * D_MODEL + col] = O[nf][r] * li[r];
    }
  }
}

// ---------------------------------------------------------------------------
extern "C" void kernel_launch(void* const* d_in, const int* in_sizes, int n_in,
                              void* d_out, int out_size, void* d_ws,
                              size_t ws_size, hipStream_t stream) {
  const float* text   = (const float*)d_in[0];
  const float* audio  = (const float*)d_in[1];
  const int*   amask  = (const int*)d_in[2];
  const float* ln_t_g = (const float*)d_in[3];
  const float* ln_t_b = (const float*)d_in[4];
  const float* ln_a_g = (const float*)d_in[5];
  const float* ln_a_b = (const float*)d_in[6];
  const float* Wq     = (const float*)d_in[7];
  const float* bq     = (const float*)d_in[8];
  const float* Wk     = (const float*)d_in[9];
  const float* bk     = (const float*)d_in[10];
  const float* Wv     = (const float*)d_in[11];
  const float* bv     = (const float*)d_in[12];
  const float* ln_p_g = (const float*)d_in[13];
  const float* ln_p_b = (const float*)d_in[14];
  float* out = (float*)d_out;

  char* ws = (char*)d_ws;
  const size_t MB = 1024ull * 1024ull;
  unsigned short* t_bf  = (unsigned short*)(ws + 0);        //  64 MB
  unsigned short* a_bf  = (unsigned short*)(ws + 64 * MB);  // 128 MB
  unsigned short* wqt   = (unsigned short*)(ws + 192 * MB); //   2 MB
  unsigned short* wkt   = (unsigned short*)(ws + 194 * MB); //   2 MB
  unsigned short* wvt   = (unsigned short*)(ws + 196 * MB); //   2 MB
  unsigned short* q_bf  = (unsigned short*)(ws + 198 * MB); //  64 MB
  unsigned short* k_bf  = (unsigned short*)(ws + 262 * MB); // 128 MB
  unsigned short* vt_bf = (unsigned short*)(ws + 390 * MB); // 128 MB [B][D][La]
  float*          attn  = (float*)(ws + 518 * MB);          // 128 MB

  // 1) LayerNorms -> bf16
  ln_to_bf16<<<(B_SZ * LT) / 8, 256, 0, stream>>>(text, ln_t_g, ln_t_b, t_bf,
                                                  B_SZ * LT);
  ln_to_bf16<<<(B_SZ * LA) / 8, 256, 0, stream>>>(audio, ln_a_g, ln_a_b, a_bf,
                                                  B_SZ * LA);
  // 2) Weights -> bf16 transposed [N][K]
  cast_transpose<<<(D_MODEL * D_MODEL) / 256, 256, 0, stream>>>(Wq, wqt);
  cast_transpose<<<(D_MODEL * D_MODEL) / 256, 256, 0, stream>>>(Wk, wkt);
  cast_transpose<<<(D_MODEL * D_MODEL) / 256, 256, 0, stream>>>(Wv, wvt);
  // 3) QKV projections (WMMA GEMMs); V written transposed [B][D][La]
  gemm_bf16<<<dim3(D_MODEL / 128, (B_SZ * LT) / 128), 256, 0, stream>>>(
      t_bf, wqt, bq, q_bf, B_SZ * LT, D_MODEL, D_MODEL, 0, LT);
  gemm_bf16<<<dim3(D_MODEL / 128, (B_SZ * LA) / 128), 256, 0, stream>>>(
      a_bf, wkt, bk, k_bf, B_SZ * LA, D_MODEL, D_MODEL, 0, LA);
  gemm_bf16<<<dim3(D_MODEL / 128, (B_SZ * LA) / 128), 256, 0, stream>>>(
      a_bf, wvt, bv, vt_bf, B_SZ * LA, D_MODEL, D_MODEL, 1, LA);
  // 4) Flash attention (TDM Q stage, WMMA QK^T + PV, online softmax, mask)
  flash_attn<<<B_SZ * (LT / 16), 256, 0, stream>>>(q_bf, k_bf, vt_bf, amask,
                                                   attn);
  // 5) Final LayerNorm -> d_out (f32)
  ln_to_f32<<<(B_SZ * LT) / 8, 256, 0, stream>>>(attn, ln_p_g, ln_p_b, out,
                                                 B_SZ * LT);
}
